// mLSTMCell_36507222016551
// MI455X (gfx1250) — matrix operand order repaired and checked by reading, not compiled, not run on hardware
//
#include <hip/hip_runtime.h>
#include <hip/hip_bf16.h>

// Problem sizes (fixed by the reference)
constexpr int Bc   = 2048;
constexpr int INc  = 4096;
constexpr int HIDc = 4096;
constexpr int Dc   = 16;
constexpr int Hc   = 256;   // HID / D

typedef __attribute__((ext_vector_type(8)))  __bf16 v8bf;
typedef __attribute__((ext_vector_type(16))) __bf16 v16bf;
typedef __attribute__((ext_vector_type(8)))  float  v8f;

// ---------------------------------------------------------------------------
// fp32 -> bf16 conversion (grid-stride, 4-wide)
// ---------------------------------------------------------------------------
__global__ void cvt_f32_bf16(const float* __restrict__ s, __bf16* __restrict__ d, long n)
{
    long i      = (long)blockIdx.x * blockDim.x + threadIdx.x;
    long stride = (long)gridDim.x * blockDim.x;
    for (long p = i * 4; p < n; p += stride * 4) {
        float4 f = *(const float4*)(s + p);
        d[p + 0] = (__bf16)f.x;
        d[p + 1] = (__bf16)f.y;
        d[p + 2] = (__bf16)f.z;
        d[p + 3] = (__bf16)f.w;
    }
}

// ---------------------------------------------------------------------------
// Async global -> LDS copy (CDNA5 async engine, tracked by ASYNCcnt).
// GVS addressing: mem = SGPR base + 32-bit VGPR offset. One instruction copies
// 16 bytes per lane to a per-lane LDS byte address.
// ---------------------------------------------------------------------------
__device__ __forceinline__ void async_copy_b128(unsigned lds_off, unsigned g_off,
                                                const __bf16* base)
{
    asm volatile("global_load_async_to_lds_b128 %0, %1, %2"
                 :: "v"(lds_off), "v"(g_off),
                    "s"((unsigned long long)(__UINTPTR_TYPE__)base)
                 : "memory");
}

// ---------------------------------------------------------------------------
// WMMA bf16 NT-GEMM:  out[M,N] = (A[M,K] @ W[N,K]^T + bias[N]) * alpha
// Block tile 128(M) x 256(N), 8 waves (2x4), wave tile 64x64 = 4x4 WMMA tiles.
// K-step 32, double-buffered LDS staging via async copies.
// LDS row stride = 40 bf16 (80B) -> conflict-free b128 reads.
// ---------------------------------------------------------------------------
constexpr int APANEL = 128 * 80;            // 10240 B
constexpr int BPANEL = 256 * 80;            // 20480 B
constexpr int STAGE  = APANEL + BPANEL;     // 30720 B
constexpr int SMEMSZ = 2 * STAGE;           // 61440 B

__global__ __launch_bounds__(256, 2)
void gemm_bf16_nt(const __bf16* __restrict__ A,
                  const __bf16* __restrict__ W,
                  const float*  __restrict__ bias,
                  float*        __restrict__ out,
                  int N, int K, float alpha)
{
    __shared__ __align__(16) char smem[SMEMSZ];

    const int tid   = threadIdx.x;
    const int lane  = tid & 31;
    const int wave  = tid >> 5;
    const int wm    = wave >> 2;            // 0..1
    const int wn    = wave & 3;             // 0..3
    const int mBlk  = blockIdx.y * 128;
    const int nBlk  = blockIdx.x * 256;
    const int half  = lane >> 4;            // 0 or 1
    const int l16   = lane & 15;

    const unsigned smemBase = (unsigned)(__UINTPTR_TYPE__)&smem[0];

    v8f acc[4][4];
#pragma unroll
    for (int i = 0; i < 4; ++i)
#pragma unroll
        for (int j = 0; j < 4; ++j)
            acc[i][j] = (v8f)(0.0f);

    // --- stage issue: copy A[128 x 32] and W[256 x 32] panels into LDS ---
    auto issue_stage = [&](int kk, int buf) {
        const unsigned aLds = smemBase + (unsigned)buf * STAGE;
        const unsigned bLds = aLds + APANEL;
#pragma unroll
        for (int j = 0; j < 2; ++j) {               // A: 512 chunks / 256 thr
            const int ch  = tid + j * 256;
            const int row = ch >> 2;
            const int c   = ch & 3;
            const unsigned g = (unsigned)(((mBlk + row) * K + kk + c * 8) * 2);
            async_copy_b128(aLds + (unsigned)(row * 80 + c * 16), g, A);
        }
#pragma unroll
        for (int j = 0; j < 4; ++j) {               // B: 1024 chunks / 256 thr
            const int ch  = tid + j * 256;
            const int row = ch >> 2;
            const int c   = ch & 3;
            const unsigned g = (unsigned)(((nBlk + row) * K + kk + c * 8) * 2);
            async_copy_b128(bLds + (unsigned)(row * 80 + c * 16), g, W);
        }
    };

    const int T = K / 32;
    issue_stage(0, 0);

    for (int t = 0; t < T; ++t) {
        const int cur = t & 1;
        if (t + 1 < T) {
            issue_stage((t + 1) * 32, (t + 1) & 1);
            // previous stage (6 copies/wave) complete; next stage in flight
            asm volatile("s_wait_asynccnt 0x6" ::: "memory");
        } else {
            asm volatile("s_wait_asynccnt 0x0" ::: "memory");
        }
        __syncthreads();

        // --- load fragments from LDS and multiply ---
        const unsigned aBase = (unsigned)cur * STAGE;
        const unsigned bBase = aBase + APANEL;
        v16bf a[4], b[4];
#pragma unroll
        for (int i = 0; i < 4; ++i) {
            const unsigned off = aBase + (unsigned)((wm * 64 + i * 16 + l16) * 80 + half * 16);
            v8bf lo = *(const v8bf*)(smem + off);
            v8bf hi = *(const v8bf*)(smem + off + 32);
            v16bf tf;
#pragma unroll
            for (int e = 0; e < 8; ++e) { tf[e] = lo[e]; tf[e + 8] = hi[e]; }
            a[i] = tf;
        }
#pragma unroll
        for (int j = 0; j < 4; ++j) {
            const unsigned off = bBase + (unsigned)((wn * 64 + j * 16 + l16) * 80 + half * 16);
            v8bf lo = *(const v8bf*)(smem + off);
            v8bf hi = *(const v8bf*)(smem + off + 32);
            v16bf tf;
#pragma unroll
            for (int e = 0; e < 8; ++e) { tf[e] = lo[e]; tf[e + 8] = hi[e]; }
            b[j] = tf;
        }
#pragma unroll
        for (int i = 0; i < 4; ++i)
#pragma unroll
            for (int j = 0; j < 4; ++j)
                acc[i][j] = __builtin_amdgcn_wmma_f32_16x16x32_bf16(
                    false, a[i], false, b[j], (short)0, acc[i][j], false, false);

        __syncthreads();   // all waves done reading before buffer is reused
    }

    // --- epilogue: bias, scale, store ---
#pragma unroll
    for (int i = 0; i < 4; ++i)
#pragma unroll
        for (int j = 0; j < 4; ++j) {
            const int colBase = nBlk + wn * 64 + j * 16 + l16;
            const float bcol  = bias[colBase];
#pragma unroll
            for (int r = 0; r < 8; ++r) {
                const int row = mBlk + wm * 64 + i * 16 + half * 8 + r;
                out[(size_t)row * N + colBase] = (acc[i][j][r] + bcol) * alpha;
            }
        }
}

// ---------------------------------------------------------------------------
// Per-head mLSTM state update. 16 lanes per head (e = column index).
// Streams C (512MB) in / C_new out fully coalesced; half-wave shuffle
// reduction for denom. Writes h_pre (bf16) for the final Wo GEMM.
// ---------------------------------------------------------------------------
__global__ __launch_bounds__(256)
void head_update(const float* __restrict__ C_in,
                 const float* __restrict__ n_in,
                 const float* __restrict__ m_in,
                 const float* __restrict__ q,
                 const float* __restrict__ k,
                 const float* __restrict__ v,
                 const float* __restrict__ it,
                 const float* __restrict__ ft,
                 float* __restrict__ C_out,
                 float* __restrict__ n_out,
                 float* __restrict__ m_out,
                 __bf16* __restrict__ hpre)
{
    const long tid = (long)blockIdx.x * blockDim.x + threadIdx.x;
    const long g   = tid >> 4;        // head index in [0, B*H)
    const int  e   = (int)(tid & 15); // column within head

    const float itl = it[g];
    const float ftl = ft[g];
    const float mo  = m_in[g];
    const float mn  = fmaxf(ftl + mo, itl);
    const float fg  = expf(ftl + mo - mn);
    const float ig  = expf(itl - mn);

    const size_t vb = (size_t)g * 16;
    const float ke  = k[vb + e];
    const float qe  = q[vb + e];
    const float nn  = fg * n_in[vb + e] + ig * ke;
    n_out[vb + e] = nn;

    const size_t cb = (size_t)g * 256;
    float Cq = 0.0f;
#pragma unroll
    for (int d = 0; d < 16; ++d) {
        const float cv = C_in[cb + d * 16 + e];
        const float vd = v[vb + d];
        const float qd = q[vb + d];
        const float cn = fg * cv + ig * vd * ke;
        C_out[cb + d * 16 + e] = cn;
        Cq = fmaf(cn, qd, Cq);
    }

    float dot = nn * qe;
#pragma unroll
    for (int off = 8; off > 0; off >>= 1)
        dot += __shfl_xor(dot, off, 16);
    const float denom = fmaxf(fabsf(dot), 1.0f);

    hpre[vb + e] = (__bf16)(Cq / denom);
    if (e == 0) m_out[g] = mn;
}

// ---------------------------------------------------------------------------
extern "C" void kernel_launch(void* const* d_in, const int* in_sizes, int n_in,
                              void* d_out, int out_size, void* d_ws, size_t ws_size,
                              hipStream_t stream)
{
    const float* x   = (const float*)d_in[0];
    const float* C   = (const float*)d_in[1];
    const float* n   = (const float*)d_in[2];
    const float* m   = (const float*)d_in[3];
    const float* Wq  = (const float*)d_in[4];
    const float* bq  = (const float*)d_in[5];
    const float* Wk  = (const float*)d_in[6];
    const float* bk  = (const float*)d_in[7];
    const float* Wv  = (const float*)d_in[8];
    const float* bv  = (const float*)d_in[9];
    const float* Wo  = (const float*)d_in[10];
    const float* bo  = (const float*)d_in[11];
    const float* Wi  = (const float*)d_in[12];
    const float* bi  = (const float*)d_in[13];
    const float* Wf  = (const float*)d_in[14];
    const float* bfg = (const float*)d_in[15];

    // ---- workspace carve-out ----
    char* ws = (char*)d_ws;
    auto carve = [&](size_t bytes) {
        void* p = (void*)ws;
        ws += (bytes + 255) & ~(size_t)255;
        return p;
    };
    __bf16* xb   = (__bf16*)carve((size_t)Bc * INc * sizeof(__bf16));
    __bf16* wqb  = (__bf16*)carve((size_t)HIDc * INc * sizeof(__bf16));
    __bf16* wkb  = (__bf16*)carve((size_t)HIDc * INc * sizeof(__bf16));
    __bf16* wvb  = (__bf16*)carve((size_t)HIDc * INc * sizeof(__bf16));
    __bf16* wob  = (__bf16*)carve((size_t)HIDc * HIDc * sizeof(__bf16));
    __bf16* wib  = (__bf16*)carve((size_t)Hc * INc * sizeof(__bf16));
    __bf16* wfb  = (__bf16*)carve((size_t)Hc * INc * sizeof(__bf16));
    float*  qf   = (float*)carve((size_t)Bc * HIDc * sizeof(float));
    float*  kf   = (float*)carve((size_t)Bc * HIDc * sizeof(float));
    float*  vf   = (float*)carve((size_t)Bc * HIDc * sizeof(float));
    float*  itf  = (float*)carve((size_t)Bc * Hc * sizeof(float));
    float*  ftf  = (float*)carve((size_t)Bc * Hc * sizeof(float));
    __bf16* hpre = (__bf16*)carve((size_t)Bc * HIDc * sizeof(__bf16));

    // ---- output layout (tuple concatenated flat) ----
    float* out_h = (float*)d_out;
    float* out_C = out_h + (size_t)Bc * HIDc;
    float* out_n = out_C + (size_t)Bc * Hc * Dc * Dc;
    float* out_m = out_n + (size_t)Bc * Hc * Dc;

    // ---- 1) fp32 -> bf16 conversions ----
    auto cvt = [&](const float* s, __bf16* d, long cnt) {
        cvt_f32_bf16<<<2048, 256, 0, stream>>>(s, d, cnt);
    };
    cvt(x,  xb,  (long)Bc * INc);
    cvt(Wq, wqb, (long)HIDc * INc);
    cvt(Wk, wkb, (long)HIDc * INc);
    cvt(Wv, wvb, (long)HIDc * INc);
    cvt(Wo, wob, (long)HIDc * HIDc);
    cvt(Wi, wib, (long)Hc * INc);
    cvt(Wf, wfb, (long)Hc * INc);

    const float inv_sqrt_d = 0.25f;   // 1/sqrt(16)

    // ---- 2) projections via WMMA GEMMs ----
    dim3 blk(256);
    dim3 grid_big(HIDc / 256, Bc / 128);   // (16, 16)
    dim3 grid_sm(Hc / 256, Bc / 128);      // (1, 16)
    gemm_bf16_nt<<<grid_big, blk, 0, stream>>>(xb, wqb, bq,  qf,  HIDc, INc, 1.0f);
    gemm_bf16_nt<<<grid_big, blk, 0, stream>>>(xb, wkb, bk,  kf,  HIDc, INc, inv_sqrt_d);
    gemm_bf16_nt<<<grid_big, blk, 0, stream>>>(xb, wvb, bv,  vf,  HIDc, INc, 1.0f);
    gemm_bf16_nt<<<grid_sm,  blk, 0, stream>>>(xb, wib, bi,  itf, Hc,   INc, 1.0f);
    gemm_bf16_nt<<<grid_sm,  blk, 0, stream>>>(xb, wfb, bfg, ftf, Hc,   INc, 1.0f);

    // ---- 3) per-head state update (streams C) ----
    {
        long total = (long)Bc * Hc * 16;            // one thread per (head, column)
        int  blocks = (int)(total / 256);           // 32768
        head_update<<<blocks, 256, 0, stream>>>(C, n, m, qf, kf, vf, itf, ftf,
                                                out_C, out_n, out_m, hpre);
    }

    // ---- 4) output projection ----
    gemm_bf16_nt<<<grid_big, blk, 0, stream>>>(hpre, wob, bo, out_h, HIDc, HIDc, 1.0f);
}
